// RWKV_40157944218042
// MI455X (gfx1250) — matrix operand order repaired and checked by reading, not compiled, not run on hardware
//
#include <hip/hip_runtime.h>
#include <hip/hip_bf16.h>
#include <math.h>

// ---------------- model constants (match reference) ----------------
#define L_   4
#define T_   1024
#define C_   768
#define HS_  64
#define H_   12          // C_/HS_
#define FF_  3072
#define V_   32000
#define DD_  64
#define DA_  64
#define DMV_ 32
#define DG_  128

typedef __attribute__((ext_vector_type(16))) __bf16 v16bf;
typedef __attribute__((ext_vector_type(4)))  __bf16 v4bf;
typedef __attribute__((ext_vector_type(2)))  __bf16 v2bf;
typedef __attribute__((ext_vector_type(8)))  float  v8f;
typedef __attribute__((ext_vector_type(4)))  float  vf4;

// =====================================================================
// GEMM: C(MxN) = A(MxK) @ B(KxN), f32 in/out, bf16 WMMA compute.
// Block = 256 threads = 8 waves; block tile 128x64, K-step 32.
// Waves arranged 4x2; each wave owns a 32x32 quadrant = 2x2
// v_wmma_f32_16x16x32_bf16 fragments. LDS double-buffered so next
// tile's global loads overlap current tile's WMMAs (1 barrier/K-step).
// Requires: M % 128 == 0, K % 32 == 0, N % 4 == 0 (N guarded).
// =====================================================================
__global__ __launch_bounds__(256) void gemm_wmma(
    const float* __restrict__ A, const float* __restrict__ B,
    float* __restrict__ C, int M, int N, int K)
{
    __shared__ __align__(16) __bf16 As[2][128][40];  // [buf][m][k], 80B rows (16B aligned)
    __shared__ __align__(16) __bf16 Bs[2][64][40];   // [buf][n][k]  (B transposed)

    const int tid  = threadIdx.x;
    const int lane = tid & 31;
    const int wave = tid >> 5;
    const int wr   = wave >> 1;       // 0..3 row quadrant
    const int wc   = wave & 1;        // 0..1 col quadrant
    const int mBase = blockIdx.y * 128;
    const int nBase = blockIdx.x * 64;

    const int mg  = lane & 15;        // 0..15
    const int grp = lane >> 4;        // 0 or 1

    // B staging: thread owns a 2(k) x 4(n) micro-block of the 32x64 tile
    const int bk = (tid >> 4) * 2;    // 0,2,..,30
    const int bn = (tid & 15) * 4;    // 0,4,..,60

    const v8f vzero = {0.f,0.f,0.f,0.f,0.f,0.f,0.f,0.f};
    v8f acc[2][2];
    acc[0][0] = vzero; acc[0][1] = vzero; acc[1][0] = vzero; acc[1][1] = vzero;

    vf4 ra[4], rb0, rb1;

    auto loadG = [&](int k0) {
        // A tile 128x32 : 4096 elems / 256 thr = 4x float4 each
        #pragma unroll
        for (int it = 0; it < 4; ++it) {
            int lin = it * 1024 + tid * 4;
            int row = lin >> 5;                 // /32
            int col = lin & 31;
            ra[it] = *(const vf4*)(A + (size_t)(mBase + row) * K + k0 + col);
        }
        // B tile 32x64
        rb0 = (vf4){0.f,0.f,0.f,0.f};
        rb1 = (vf4){0.f,0.f,0.f,0.f};
        int n = nBase + bn;
        if (n < N) {                            // N%4==0 -> whole float4 in/out
            rb0 = *(const vf4*)(B + (size_t)(k0 + bk    ) * N + n);
            rb1 = *(const vf4*)(B + (size_t)(k0 + bk + 1) * N + n);
        }
    };
    auto storeL = [&](int buf) {
        #pragma unroll
        for (int it = 0; it < 4; ++it) {
            int lin = it * 1024 + tid * 4;
            int row = lin >> 5;
            int col = lin & 31;
            v4bf pk;
            pk[0] = (__bf16)ra[it].x; pk[1] = (__bf16)ra[it].y;
            pk[2] = (__bf16)ra[it].z; pk[3] = (__bf16)ra[it].w;
            *(v4bf*)&As[buf][row][col] = pk;    // one ds_store_b64
        }
        #pragma unroll
        for (int j = 0; j < 4; ++j) {
            v2bf pk;
            pk[0] = (__bf16)rb0[j];             // k = bk
            pk[1] = (__bf16)rb1[j];             // k = bk+1
            *(v2bf*)&Bs[buf][bn + j][bk] = pk;  // one ds_store_b32
        }
    };

    const int nsteps = K >> 5;
    loadG(0);
    storeL(0);
    __syncthreads();

    for (int s = 0; s < nsteps; ++s) {
        const int cur = s & 1;
        if (s + 1 < nsteps) loadG((s + 1) << 5);   // prefetch (waits land after WMMAs)

        // ---- fragments in contiguous k order k = 16*grp + e (same for A and B) ----
        // A 16x32: lanes 0-15 m=lane K=0..15; lanes 16-31 K=16..31 (matches B)
        v16bf af[2];
        #pragma unroll
        for (int sm = 0; sm < 2; ++sm) {
            const __bf16* pa = &As[cur][wr*32 + sm*16 + mg][0];
            #pragma unroll
            for (int e = 0; e < 8; ++e) {
                af[sm][e]   = pa[grp*16 + e];
                af[sm][e+8] = pa[grp*16 + 8 + e];
            }
        }
        // B 32x16: VGPR v holds K=2v,2v+1; lanes 0-15 n=lane K=0..15; lanes 16-31 K=16..31
        v16bf bfv[2];
        #pragma unroll
        for (int sn = 0; sn < 2; ++sn) {
            const __bf16* pb = &Bs[cur][wc*32 + sn*16 + mg][0];
            #pragma unroll
            for (int e = 0; e < 16; ++e)
                bfv[sn][e] = pb[grp*16 + e];
        }
        #pragma unroll
        for (int sm = 0; sm < 2; ++sm)
            #pragma unroll
            for (int sn = 0; sn < 2; ++sn)
                acc[sm][sn] = __builtin_amdgcn_wmma_f32_16x16x32_bf16(
                    false, af[sm], false, bfv[sn], (short)0, acc[sm][sn], false, false);

        if (s + 1 < nsteps) storeL(cur ^ 1);
        __syncthreads();
    }

    // ---- epilogue: C/D layout: elem v -> row m0 + v + 8*grp, col n0 + (lane&15) ----
    #pragma unroll
    for (int sm = 0; sm < 2; ++sm)
        #pragma unroll
        for (int sn = 0; sn < 2; ++sn) {
            int col = nBase + wc*32 + sn*16 + mg;
            if (col < N) {
                #pragma unroll
                for (int v = 0; v < 8; ++v) {
                    int row = mBase + wr*32 + sm*16 + v + 8*grp;
                    C[(size_t)row * N + col] = acc[sm][sn][v];
                }
            }
        }
}

// =====================================================================
// RWKV-7 serial scan. One block per head, 64 threads; thread i holds
// state row S[i][0..63] in registers (fully unrolled).
//   sa_i = sum_j S[i][j] * (-kk_j)
//   S[i][j] = S[i][j]*dec_j + sa_i*(kk_j*a_j) + v_i*k_j
//   y_i = sum_j S[i][j]*r_j
// =====================================================================
__global__ __launch_bounds__(64) void k_wkv(
    const float* __restrict__ r, const float* __restrict__ dec,
    const float* __restrict__ k, const float* __restrict__ v,
    const float* __restrict__ kk, const float* __restrict__ a,
    float* __restrict__ y)
{
    const int h = blockIdx.x;
    const int i = threadIdx.x;

    float S[HS_];
    #pragma unroll
    for (int j = 0; j < HS_; ++j) S[j] = 0.f;

    __shared__ float sr[HS_], sd[HS_], sk[HS_], skk[HS_], sa[HS_];

    for (int t = 0; t < T_; ++t) {
        const size_t base = (size_t)t * C_ + h * HS_;
        sr[i]  = r[base + i];
        sd[i]  = dec[base + i];
        sk[i]  = k[base + i];
        skk[i] = kk[base + i];
        sa[i]  = a[base + i];
        float vi = v[base + i];
        __syncthreads();

        float sa_i = 0.f;
        #pragma unroll
        for (int j = 0; j < HS_; ++j) sa_i -= S[j] * skk[j];

        float yi = 0.f;
        #pragma unroll
        for (int j = 0; j < HS_; ++j) {
            S[j] = S[j] * sd[j] + sa_i * (skk[j] * sa[j]) + vi * sk[j];
            yi  += S[j] * sr[j];
        }
        y[base + i] = yi;
        __syncthreads();
    }
}

// =====================================================================
// small kernels
// =====================================================================
__global__ __launch_bounds__(256) void k_embed(
    const int* __restrict__ idx, const float* __restrict__ emb, float* __restrict__ x)
{
    int i = blockIdx.x * 256 + threadIdx.x;
    if (i < T_ * C_) {
        int t = i / C_, c = i % C_;
        x[i] = emb[(size_t)idx[t] * C_ + c];
    }
}

// block per row; in-place safe (one block owns the row)
__global__ __launch_bounds__(256) void k_layernorm(
    const float* __restrict__ xin, const float* __restrict__ g, const float* __restrict__ b,
    float* __restrict__ out, float eps)
{
    const int row = blockIdx.x;
    const int tid = threadIdx.x;
    const float* xr = xin + (size_t)row * C_;
    float vals[3], lsum = 0.f, lsq = 0.f;
    #pragma unroll
    for (int j = 0; j < 3; ++j) {
        float vv = xr[tid + j * 256];
        vals[j] = vv; lsum += vv; lsq += vv * vv;
    }
    __shared__ float s1[256], s2[256];
    s1[tid] = lsum; s2[tid] = lsq;
    __syncthreads();
    for (int s = 128; s > 0; s >>= 1) {
        if (tid < s) { s1[tid] += s1[tid + s]; s2[tid] += s2[tid + s]; }
        __syncthreads();
    }
    float m   = s1[0] * (1.f / C_);
    float var = s2[0] * (1.f / C_) - m * m;
    float inv = rsqrtf(var + eps);
    #pragma unroll
    for (int j = 0; j < 3; ++j) {
        int c = tid + j * 256;
        out[(size_t)row * C_ + c] = (vals[j] - m) * inv * g[c] + b[c];
    }
}

// out = xn + (shift(xn) - xn) * mix[c]
__global__ __launch_bounds__(256) void k_shiftmix(
    const float* __restrict__ xn, const float* __restrict__ mixv, float* __restrict__ out)
{
    int i = blockIdx.x * 256 + threadIdx.x;
    if (i < T_ * C_) {
        int t = i / C_, c = i % C_;
        float cur = xn[i];
        float prev = (t > 0) ? xn[i - C_] : 0.f;
        out[i] = cur + (prev - cur) * mixv[c];
    }
}

__global__ __launch_bounds__(256) void k_tanh(float* __restrict__ x, int n) {
    int i = blockIdx.x * 256 + threadIdx.x;
    if (i < n) x[i] = tanhf(x[i]);
}
__global__ __launch_bounds__(256) void k_sigmoid(float* __restrict__ x, int n) {
    int i = blockIdx.x * 256 + threadIdx.x;
    if (i < n) x[i] = 1.f / (1.f + expf(-x[i]));
}
__global__ __launch_bounds__(256) void k_relusq(float* __restrict__ x, int n) {
    int i = blockIdx.x * 256 + threadIdx.x;
    if (i < n) { float v = fmaxf(x[i], 0.f); x[i] = v * v; }
}
__global__ __launch_bounds__(256) void k_add(float* __restrict__ x, const float* __restrict__ y, int n) {
    int i = blockIdx.x * 256 + threadIdx.x;
    if (i < n) x[i] += y[i];
}
__global__ __launch_bounds__(256) void k_mul(
    const float* __restrict__ a, const float* __restrict__ b, float* __restrict__ out, int n) {
    int i = blockIdx.x * 256 + threadIdx.x;
    if (i < n) out[i] = a[i] * b[i];
}
// dec = exp(-exp(-softplus(-(w0+wt)) - 0.5))
__global__ __launch_bounds__(256) void k_wact(
    const float* __restrict__ wt, const float* __restrict__ w0, float* __restrict__ dec) {
    int i = blockIdx.x * 256 + threadIdx.x;
    if (i < T_ * C_) {
        int c = i % C_;
        float z = w0[c] + wt[i];
        float w = -log1pf(expf(-z)) - 0.5f;
        dec[i] = expf(-expf(w));
    }
}
// a = sigmoid(a0 + at)
__global__ __launch_bounds__(256) void k_aact(
    const float* __restrict__ at, const float* __restrict__ a0, float* __restrict__ a) {
    int i = blockIdx.x * 256 + threadIdx.x;
    if (i < T_ * C_) {
        int c = i % C_;
        a[i] = 1.f / (1.f + expf(-(a0[c] + at[i])));
    }
}
// v = v + (vfirst - v) * sigmoid(v0 + vt)
__global__ __launch_bounds__(256) void k_vblend(
    float* __restrict__ v, const float* __restrict__ vfirst,
    const float* __restrict__ vt, const float* __restrict__ v0) {
    int i = blockIdx.x * 256 + threadIdx.x;
    if (i < T_ * C_) {
        int c = i % C_;
        float s = 1.f / (1.f + expf(-(v0[c] + vt[i])));
        v[i] = v[i] + (vfirst[i] - v[i]) * s;
    }
}
// k = k * (1 + (a-1)*k_a)
__global__ __launch_bounds__(256) void k_kmod(
    float* __restrict__ k, const float* __restrict__ a, const float* __restrict__ ka) {
    int i = blockIdx.x * 256 + threadIdx.x;
    if (i < T_ * C_) {
        int c = i % C_;
        k[i] = k[i] * (1.f + (a[i] - 1.f) * ka[c]);
    }
}
// kk = normalize(k * k_k) per (t,h) over HS
__global__ __launch_bounds__(64) void k_kknorm(
    const float* __restrict__ k, const float* __restrict__ kkp, float* __restrict__ kk)
{
    int th = blockIdx.x;
    int t = th / H_, h = th % H_;
    int i = threadIdx.x;
    int c = h * HS_ + i;
    size_t off = (size_t)t * C_ + c;
    float v = k[off] * kkp[c];
    __shared__ float s1[64];
    s1[i] = v * v;
    __syncthreads();
    for (int s = 32; s > 0; s >>= 1) { if (i < s) s1[i] += s1[i + s]; __syncthreads(); }
    float nrm = sqrtf(s1[0]);
    kk[off] = v / fmaxf(nrm, 1e-12f);
}
// per-head GroupNorm(eps=6.4e-4) with affine + bonus: y += sum(r*k*r_k)*v
__global__ __launch_bounds__(64) void k_gnorm_rk(
    float* __restrict__ y, const float* __restrict__ r, const float* __restrict__ k,
    const float* __restrict__ v, const float* __restrict__ lnxg,
    const float* __restrict__ lnxb, const float* __restrict__ rk)
{
    int th = blockIdx.x;
    int t = th / H_, h = th % H_;
    int i = threadIdx.x;
    int c = h * HS_ + i;
    size_t off = (size_t)t * C_ + c;
    float yv = y[off];
    __shared__ float s1[64], s2[64];
    s1[i] = yv; s2[i] = yv * yv;
    __syncthreads();
    for (int s = 32; s > 0; s >>= 1) {
        if (i < s) { s1[i] += s1[i + s]; s2[i] += s2[i + s]; }
        __syncthreads();
    }
    float m   = s1[0] * (1.f / HS_);
    float var = s2[0] * (1.f / HS_) - m * m;
    __syncthreads();
    yv = (yv - m) * rsqrtf(var + 0.00064f) * lnxg[c] + lnxb[c];
    s1[i] = r[off] * k[off] * rk[c];
    __syncthreads();
    for (int s = 32; s > 0; s >>= 1) { if (i < s) s1[i] += s1[i + s]; __syncthreads(); }
    yv += s1[0] * v[off];
    y[off] = yv;
}

// =====================================================================
// host orchestration
// =====================================================================
static inline void gemm(const float* A, const float* B, float* Cp,
                        int M, int N, int K, hipStream_t s) {
    dim3 grid((N + 63) / 64, (M + 127) / 128);
    gemm_wmma<<<grid, 256, 0, s>>>(A, B, Cp, M, N, K);
}

extern "C" void kernel_launch(void* const* d_in, const int* in_sizes, int n_in,
                              void* d_out, int out_size, void* d_ws, size_t ws_size,
                              hipStream_t stream) {
    // inputs in setup_inputs() dict order, params dict flattened in key order
    int p = 0;
    const int*   idx     = (const int*)d_in[p++];     // (B,T)
    const float* emb     = (const float*)d_in[p++];   // (V,C)
    const float* ln0_g   = (const float*)d_in[p++];
    const float* ln0_b   = (const float*)d_in[p++];
    const float* ln1_g   = (const float*)d_in[p++];
    const float* ln1_b   = (const float*)d_in[p++];
    const float* ln2_g   = (const float*)d_in[p++];
    const float* ln2_b   = (const float*)d_in[p++];
    const float* x_mix   = (const float*)d_in[p++];   // (L,6,C)
    const float* w0      = (const float*)d_in[p++];
    const float* w1      = (const float*)d_in[p++];
    const float* w2      = (const float*)d_in[p++];
    const float* a0      = (const float*)d_in[p++];
    const float* a1      = (const float*)d_in[p++];
    const float* a2      = (const float*)d_in[p++];
    const float* v0      = (const float*)d_in[p++];
    const float* v1      = (const float*)d_in[p++];
    const float* v2      = (const float*)d_in[p++];
    const float* g1      = (const float*)d_in[p++];
    const float* g2      = (const float*)d_in[p++];
    const float* k_k     = (const float*)d_in[p++];
    const float* k_a     = (const float*)d_in[p++];
    const float* r_k     = (const float*)d_in[p++];   // (L,H,HS) == (L,C)
    const float* Wr      = (const float*)d_in[p++];
    const float* Wk      = (const float*)d_in[p++];
    const float* Wv      = (const float*)d_in[p++];
    const float* Wo      = (const float*)d_in[p++];
    const float* lnx_g   = (const float*)d_in[p++];
    const float* lnx_b   = (const float*)d_in[p++];
    const float* fxk     = (const float*)d_in[p++];
    const float* Wfk     = (const float*)d_in[p++];
    const float* Wfv     = (const float*)d_in[p++];
    const float* head_w  = (const float*)d_in[p++];   // (C,V)
    const float* lnout_g = (const float*)d_in[p++];
    const float* lnout_b = (const float*)d_in[p++];

    const size_t NC = (size_t)T_ * C_;
    float* ws = (float*)d_ws;
    float* x      = ws; ws += NC;
    float* xn     = ws; ws += NC;
    float* mixb   = ws; ws += NC;
    float* rb     = ws; ws += NC;
    float* kb     = ws; ws += NC;
    float* vb     = ws; ws += NC;
    float* ab     = ws; ws += NC;
    float* gb     = ws; ws += NC;
    float* decb   = ws; ws += NC;
    float* kkb    = ws; ws += NC;
    float* vfirst = ws; ws += NC;
    float* tmpC   = ws; ws += NC;
    float* yb     = ws; ws += NC;
    float* smallb = ws; ws += (size_t)T_ * DG_;       // up to 1024x128
    float* ffb    = ws; ws += (size_t)T_ * FF_;

    const int nTC = T_ * C_;
    dim3 gTC((nTC + 255) / 256);
    dim3 gFF((T_ * FF_ + 255) / 256);

    // embedding
    k_embed<<<gTC, 256, 0, stream>>>(idx, emb, x);

    for (int i = 0; i < L_; ++i) {
        const float* mix_i = x_mix + (size_t)i * 6 * C_;
        if (i == 0)
            k_layernorm<<<T_, 256, 0, stream>>>(x, ln0_g, ln0_b, x, 1e-5f);

        // ---- time mix ----
        k_layernorm<<<T_, 256, 0, stream>>>(x, ln1_g + i * C_, ln1_b + i * C_, xn, 1e-5f);

        // r
        k_shiftmix<<<gTC, 256, 0, stream>>>(xn, mix_i + 0 * C_, mixb);
        gemm(mixb, Wr + (size_t)i * C_ * C_, rb, T_, C_, C_, stream);
        // w  ->  dec
        k_shiftmix<<<gTC, 256, 0, stream>>>(xn, mix_i + 1 * C_, mixb);
        gemm(mixb, w1 + (size_t)i * C_ * DD_, smallb, T_, DD_, C_, stream);
        k_tanh<<<(T_ * DD_ + 255) / 256, 256, 0, stream>>>(smallb, T_ * DD_);
        gemm(smallb, w2 + (size_t)i * DD_ * C_, tmpC, T_, C_, DD_, stream);
        k_wact<<<gTC, 256, 0, stream>>>(tmpC, w0 + i * C_, decb);
        // k
        k_shiftmix<<<gTC, 256, 0, stream>>>(xn, mix_i + 2 * C_, mixb);
        gemm(mixb, Wk + (size_t)i * C_ * C_, kb, T_, C_, C_, stream);
        // v (+ v_first)
        k_shiftmix<<<gTC, 256, 0, stream>>>(xn, mix_i + 3 * C_, mixb);
        gemm(mixb, Wv + (size_t)i * C_ * C_, vb, T_, C_, C_, stream);
        if (i == 0) {
            hipMemcpyAsync(vfirst, vb, NC * sizeof(float), hipMemcpyDeviceToDevice, stream);
        } else {
            gemm(mixb, v1 + (size_t)i * C_ * DMV_, smallb, T_, DMV_, C_, stream);
            gemm(smallb, v2 + (size_t)i * DMV_ * C_, tmpC, T_, C_, DMV_, stream);
            k_vblend<<<gTC, 256, 0, stream>>>(vb, vfirst, tmpC, v0 + i * C_);
        }
        // a
        k_shiftmix<<<gTC, 256, 0, stream>>>(xn, mix_i + 4 * C_, mixb);
        gemm(mixb, a1 + (size_t)i * C_ * DA_, smallb, T_, DA_, C_, stream);
        gemm(smallb, a2 + (size_t)i * DA_ * C_, tmpC, T_, C_, DA_, stream);
        k_aact<<<gTC, 256, 0, stream>>>(tmpC, a0 + i * C_, ab);
        // g
        k_shiftmix<<<gTC, 256, 0, stream>>>(xn, mix_i + 5 * C_, mixb);
        gemm(mixb, g1 + (size_t)i * C_ * DG_, smallb, T_, DG_, C_, stream);
        k_sigmoid<<<(T_ * DG_ + 255) / 256, 256, 0, stream>>>(smallb, T_ * DG_);
        gemm(smallb, g2 + (size_t)i * DG_ * C_, gb, T_, C_, DG_, stream);

        // kk from pre-mod k, then modify k
        k_kknorm<<<T_ * H_, 64, 0, stream>>>(kb, k_k + i * C_, kkb);
        k_kmod<<<gTC, 256, 0, stream>>>(kb, ab, k_a + i * C_);

        // serial scan
        k_wkv<<<H_, 64, 0, stream>>>(rb, decb, kb, vb, kkb, ab, yb);

        // per-head groupnorm + bonus
        k_gnorm_rk<<<T_ * H_, 64, 0, stream>>>(yb, rb, kb, vb,
                                               lnx_g + i * C_, lnx_b + i * C_, r_k + i * C_);
        // output proj with gate, residual
        k_mul<<<gTC, 256, 0, stream>>>(yb, gb, mixb, nTC);
        gemm(mixb, Wo + (size_t)i * C_ * C_, tmpC, T_, C_, C_, stream);
        k_add<<<gTC, 256, 0, stream>>>(x, tmpC, nTC);

        // ---- channel mix ----
        k_layernorm<<<T_, 256, 0, stream>>>(x, ln2_g + i * C_, ln2_b + i * C_, xn, 1e-5f);
        k_shiftmix<<<gTC, 256, 0, stream>>>(xn, fxk + i * C_, mixb);
        gemm(mixb, Wfk + (size_t)i * C_ * FF_, ffb, T_, FF_, C_, stream);
        k_relusq<<<gFF, 256, 0, stream>>>(ffb, T_ * FF_);
        gemm(ffb, Wfv + (size_t)i * FF_ * C_, tmpC, T_, C_, FF_, stream);
        k_add<<<gTC, 256, 0, stream>>>(x, tmpC, nTC);
    }

    // final LN + head
    k_layernorm<<<T_, 256, 0, stream>>>(x, lnout_g, lnout_b, xn, 1e-5f);
    gemm(xn, head_w, (float*)d_out, T_, V_, C_, stream);
}
